// GeoModRank_72550587564542
// MI455X (gfx1250) — matrix operand order repaired and checked by reading, not compiled, not run on hardware
//
#include <hip/hip_runtime.h>
#include <hip/hip_bf16.h>
#include <math.h>

typedef __attribute__((ext_vector_type(16))) _Float16 v16h;
typedef __attribute__((ext_vector_type(8)))  float    v8f;

// ---------------------------------------------------------------------------
// A-fragment loader: 16x32 f16 A tile, CDNA5 WMMA layout.
// lanes 0-15 : row m0+lane,   VGPR0-3 = K 0..7,  VGPR4-7 = K 16..23
// lanes16-31 : row m0+lane-16,VGPR0-3 = K 8..15, VGPR4-7 = K 24..31
// ---------------------------------------------------------------------------
__device__ inline v16h gnn_load_a(const _Float16* __restrict__ A, int lda,
                                  int m0, int k0, int lane) {
  const int hs = (lane >> 4) & 1;
  const _Float16* row = A + (size_t)(m0 + (lane & 15)) * lda + (k0 + hs * 8);
  v16h a;
#pragma unroll
  for (int v = 0; v < 8; ++v) {
    const int kb = (v < 4) ? (2 * v) : (8 + 2 * v);  // 0,2,4,6,16,18,20,22
    a[2 * v]     = row[kb];
    a[2 * v + 1] = row[kb + 1];
  }
  return a;
}

// ---------------------------------------------------------------------------
// Register-resident-A dual-input WMMA GEMM:
//   C[M,Nw] = act( A1[M,KT1*32]·W1 + (KT2? A2[M,KT2*32]·W2 : 0) + biasPk )
// Each wave32 owns a 16-row strip: loads all A fragments once into VGPRs,
// then sweeps every 16-col tile loading only packed B fragments (1 b128 pair
// + 1 v_wmma per K-step). W1/W2 packed as frag(nt*KT+k): 32 lanes x 16 f16.
// biasPk is zero-padded to nTiles*16 -> unconditional per-tile load.
// M must be a multiple of 16.
// ---------------------------------------------------------------------------
template <int KT1, int KT2>
__global__ __launch_bounds__(256) void gnn_wmma_gemm(
    const _Float16* __restrict__ A1, int lda1, const _Float16* __restrict__ Wp1,
    const _Float16* __restrict__ A2, int lda2, const _Float16* __restrict__ Wp2,
    const float* __restrict__ biasPk, int act_gelu,
    float* __restrict__ C, int ldc, _Float16* __restrict__ C16, int ldc16,
    int M, int Nw, int nTiles)
{
  const int wid  = threadIdx.x >> 5;
  const int lane = threadIdx.x & 31;
  const int mt   = blockIdx.x * 8 + wid;
  if (mt >= (M >> 4)) return;                // wave-uniform exit
  const int m0 = mt << 4;

  // A fragments live in VGPRs for the whole N sweep.
  v16h a1[KT1];
#pragma unroll
  for (int k = 0; k < KT1; ++k) a1[k] = gnn_load_a(A1, lda1, m0, k * 32, lane);
  v16h a2[KT2 ? KT2 : 1];
  if (KT2) {
#pragma unroll
    for (int k = 0; k < KT2; ++k) a2[k] = gnn_load_a(A2, lda2, m0, k * 32, lane);
  }

  const int nl    = lane & 15;
  const int mbase = m0 + (((lane >> 4) & 1) ? 8 : 0);

  for (int nt = 0; nt < nTiles; ++nt) {
    const _Float16* wp = Wp1 + (size_t)nt * KT1 * 512 + (size_t)lane * 16;
    if (nt + 1 < nTiles)
      __builtin_prefetch((const void*)(wp + (size_t)KT1 * 512), 0, 3);
    const float bv = biasPk[nt * 16 + nl];   // unconditional (zero-padded)
    v8f acc = {};
#pragma unroll
    for (int k = 0; k < KT1; ++k) {
      v16h b = *(const v16h*)(wp + (size_t)k * 512);
      acc = __builtin_amdgcn_wmma_f32_16x16x32_f16(false, a1[k], false, b,
                                                   (short)0, acc, false, false);
    }
    if (KT2) {
      const _Float16* wp2 = Wp2 + (size_t)nt * KT2 * 512 + (size_t)lane * 16;
#pragma unroll
      for (int k = 0; k < KT2; ++k) {
        v16h b = *(const v16h*)(wp2 + (size_t)k * 512);
        acc = __builtin_amdgcn_wmma_f32_16x16x32_f16(false, a2[k], false, b,
                                                     (short)0, acc, false, false);
      }
    }

    // Epilogue. C/D layout: lane&15 = column, VGPR v = row (lanes>=16 -> +8).
    const int  n   = nt * 16 + nl;
    const bool nok = (n < Nw);
#pragma unroll
    for (int v = 0; v < 8; ++v) {
      float x = acc[v] + bv;
      if (act_gelu) x = 0.5f * x * (1.0f + erff(x * 0.70710678118654752f));
      if (nok) {
        const int m = mbase + v;
        if (C)   C[(size_t)m * ldc + n]     = x;
        if (C16) C16[(size_t)m * ldc16 + n] = (_Float16)x;
      }
    }
  }
}

// ---------------------------------------------------------------------------
// Pack f32 row-major W[K,Nw] into WMMA B fragments (f16), zero-padded.
// frag index = nt_idx*kt + k_idx ; per frag: 32 lanes x 16 halves.
// lane<16: col n0+lane, halves j -> K = k*32 + (j<8 ? j : j+8) ; lane>=16: K += 8.
// ---------------------------------------------------------------------------
__global__ void gnn_pack_w(const float* __restrict__ W, _Float16* __restrict__ Pk,
                           int K, int Nw, int kt, int nt) {
  const int tid = blockIdx.x * blockDim.x + threadIdx.x;
  const int total = kt * nt * 32;
  if (tid >= total) return;
  const int lane = tid & 31;
  const int frag = tid >> 5;
  const int k  = frag % kt;
  const int n  = frag / kt;
  const int n0 = n * 16 + (lane & 15);
  const int kb = k * 32 + ((lane >> 4) ? 8 : 0);
  _Float16* out = Pk + (size_t)frag * 512 + (size_t)lane * 16;
#pragma unroll
  for (int j = 0; j < 16; ++j) {
    const int kk = kb + (j < 8 ? j : j + 8);
    const float v = (kk < K && n0 < Nw) ? W[(size_t)kk * Nw + n0] : 0.0f;
    out[j] = (_Float16)v;
  }
}

// Pack bias: zero-pad to nTiles*16 floats
__global__ void gnn_pack_bias(const float* __restrict__ b, float* __restrict__ out,
                              int Nw, int total) {
  const int tid = blockIdx.x * blockDim.x + threadIdx.x;
  if (tid >= total) return;
  out[tid] = (tid < Nw) ? b[tid] : 0.0f;
}

// Convert f32 [M,D] -> f16 [M,ldy] (zero-padded columns D..ldy-1)
__global__ void gnn_cvt_in(const float* __restrict__ X, _Float16* __restrict__ Y,
                           int M, int D, int ldy) {
  const int tid = blockIdx.x * blockDim.x + threadIdx.x;
  if (tid >= M * ldy) return;
  const int m = tid / ldy, f = tid % ldy;
  Y[tid] = (f < D) ? (_Float16)X[(size_t)m * D + f] : (_Float16)0.0f;
}

// Edge scatter: agg[dst] += h16[src]  (6 chunks of 32 features per edge)
__global__ void gnn_scatter(const _Float16* __restrict__ h16,
                            const int* __restrict__ src, const int* __restrict__ dst,
                            float* __restrict__ agg, int E, int Hf) {
  const int tid = blockIdx.x * blockDim.x + threadIdx.x;
  const int chunks = Hf / 32;
  if (tid >= E * chunks) return;
  const int e = tid / chunks, c = tid % chunks;
  const _Float16* hp = h16 + (size_t)src[e] * Hf + c * 32;
  float* ap = agg + (size_t)dst[e] * Hf + c * 32;
#pragma unroll 8
  for (int i = 0; i < 32; ++i) atomicAdd(ap + i, (float)hp[i]);
}

__global__ void gnn_degree(const int* __restrict__ dst, float* __restrict__ deg, int E) {
  const int tid = blockIdx.x * blockDim.x + threadIdx.x;
  if (tid < E) atomicAdd(deg + dst[tid], 1.0f);
}

__global__ void gnn_mean(const float* __restrict__ agg, const float* __restrict__ deg,
                         _Float16* __restrict__ mean16, int Nn, int Hf) {
  const int tid = blockIdx.x * blockDim.x + threadIdx.x;
  if (tid >= Nn * Hf) return;
  float d = deg[tid / Hf];
  d = d < 1.0f ? 1.0f : d;
  mean16[tid] = (_Float16)(agg[tid] / d);
}

// ---------------------------------------------------------------------------
// Host orchestration
// ---------------------------------------------------------------------------
extern "C" void kernel_launch(void* const* d_in, const int* in_sizes, int n_in,
                              void* d_out, int out_size, void* d_ws, size_t ws_size,
                              hipStream_t stream) {
  // Logical weight order:
  //  0-3 enc_clim(W1,b1,W2,b2) 4-7 enc_poll 8-11 enc_soc 12-13 pre(W,b)
  //  14-16 conv0(Wl,bl,Wr) 17-19 conv1 20-21 post(W,b)
  //  22-25 dec_clim 26-29 dec_poll 30-33 dec_soc
  static const int idxA[34] = {4,5,6,7, 8,9,10,11, 12,13,14,15, 16,17,
                               18,19,20, 21,22,23, 24,25,
                               26,27,28,29, 30,31,32,33, 34,35,36,37};
  static const int idxB[34] = {19,21,20,22, 23,25,24,26, 27,29,28,30, 33,34,
                               1,3,2, 4,6,5, 31,32,
                               7,9,8,10, 11,13,12,14, 15,17,16,18};
  const bool sA = (in_sizes[0] == 1600000);   // insertion order vs sorted tree
  auto P = [&](int l) -> const float* { return (const float*)d_in[sA ? idxA[l] : idxB[l]]; };

  const float* xc = (const float*)d_in[sA ? 0 : 35];
  const float* xp = (const float*)d_in[sA ? 1 : 36];
  const float* xs = (const float*)d_in[sA ? 2 : 37];
  const int*   ei = (const int*)  d_in[sA ? 3 : 0];
  const int Nn = in_sizes[sA ? 0 : 35] / 32;   // 50000
  const int E  = in_sizes[sA ? 3 : 0] / 2;     // 400000
  const int H  = 192;
  const int* srcI = ei;
  const int* dstI = ei + E;

  // workspace layout
  size_t off = 0;
  auto wsAlloc = [&](size_t bytes) -> void* {
    void* p = (void*)((char*)d_ws + off);
    off += (bytes + 255) & ~(size_t)255;
    return p;
  };
  _Float16* xc16   = (_Float16*)wsAlloc((size_t)Nn * 32 * 2);
  _Float16* xp16   = (_Float16*)wsAlloc((size_t)Nn * 32 * 2);
  _Float16* xs16   = (_Float16*)wsAlloc((size_t)Nn * 32 * 2);
  _Float16* t16    = (_Float16*)wsAlloc((size_t)Nn * 128 * 2);
  _Float16* hA     = (_Float16*)wsAlloc((size_t)Nn * H * 2);
  _Float16* hB     = (_Float16*)wsAlloc((size_t)Nn * H * 2);
  _Float16* mean16 = (_Float16*)wsAlloc((size_t)Nn * H * 2);  // reused as z16
  float*    agg    = (float*)   wsAlloc((size_t)Nn * H * 4);
  float*    deg    = (float*)   wsAlloc((size_t)Nn * 4);
  _Float16* wpool  = (_Float16*)wsAlloc((size_t)1 << 20);
  float*    bpool  = (float*)   wsAlloc((size_t)1 << 15);

  // weight packing
  size_t wo = 0;
  auto pack = [&](int lw, int K, int Nw, int* ktOut) -> _Float16* {
    const int kt = (K + 31) / 32, nt = (Nw + 15) / 16;
    _Float16* dst = wpool + wo;
    wo += (size_t)kt * nt * 512;
    const int tot = kt * nt * 32;
    gnn_pack_w<<<(tot + 255) / 256, 256, 0, stream>>>(P(lw), dst, K, Nw, kt, nt);
    *ktOut = kt;
    return dst;
  };
  // bias packing (zero-padded to tile multiple)
  size_t bo = 0;
  auto packB = [&](int lb, int Nw) -> float* {
    const int tot = ((Nw + 15) / 16) * 16;
    float* dst = bpool + bo;
    bo += (size_t)tot;
    gnn_pack_bias<<<(tot + 255) / 256, 256, 0, stream>>>(P(lb), dst, Nw, tot);
    return dst;
  };

  int ktEC1, ktEC2, ktEP1, ktEP2, ktES1, ktES2, ktPre;
  int ktWl0, ktWr0, ktWl1, ktWr1, ktPost;
  int ktDC1, ktDC2, ktDP1, ktDP2, ktDS1, ktDS2;
  _Float16* wEC1 = pack(0, 32, 128, &ktEC1);   _Float16* wEC2 = pack(2, 128, 64, &ktEC2);
  _Float16* wEP1 = pack(4, 16, 128, &ktEP1);   _Float16* wEP2 = pack(6, 128, 64, &ktEP2);
  _Float16* wES1 = pack(8, 24, 128, &ktES1);   _Float16* wES2 = pack(10, 128, 64, &ktES2);
  _Float16* wPre = pack(12, H, H, &ktPre);
  _Float16* wWl0 = pack(14, H, H, &ktWl0);     _Float16* wWr0 = pack(16, H, H, &ktWr0);
  _Float16* wWl1 = pack(17, H, H, &ktWl1);     _Float16* wWr1 = pack(19, H, H, &ktWr1);
  _Float16* wPost = pack(20, H, H, &ktPost);
  _Float16* wDC1 = pack(22, 64, 128, &ktDC1);  _Float16* wDC2 = pack(24, 128, 32, &ktDC2);
  _Float16* wDP1 = pack(26, 64, 128, &ktDP1);  _Float16* wDP2 = pack(28, 128, 16, &ktDP2);
  _Float16* wDS1 = pack(30, 64, 128, &ktDS1);  _Float16* wDS2 = pack(32, 128, 24, &ktDS2);

  float* bEC1 = packB(1, 128);  float* bEC2 = packB(3, 64);
  float* bEP1 = packB(5, 128);  float* bEP2 = packB(7, 64);
  float* bES1 = packB(9, 128);  float* bES2 = packB(11, 64);
  float* bPre = packB(13, H);
  float* bBl0 = packB(15, H);   float* bBl1 = packB(18, H);
  float* bPost = packB(21, H);
  float* bDC1 = packB(23, 128); float* bDC2 = packB(25, 32);
  float* bDP1 = packB(27, 128); float* bDP2 = packB(29, 16);
  float* bDS1 = packB(31, 128); float* bDS2 = packB(33, 24);

  // templated GEMM dispatch (KT configurations used by this model)
  auto gemm = [&](const _Float16* A1, int lda1, const _Float16* W1, int kt1,
                  const _Float16* A2, int lda2, const _Float16* W2, int kt2,
                  const float* biasPk, int act, float* C, int ldc,
                  _Float16* C16, int ldc16, int M, int Nw) {
    const int nT = (Nw + 15) / 16;
    const int blocks = ((M / 16) + 7) / 8;
    if (kt2 == 0) {
      switch (kt1) {
        case 1:
          gnn_wmma_gemm<1, 0><<<blocks, 256, 0, stream>>>(
              A1, lda1, W1, nullptr, 0, nullptr, biasPk, act, C, ldc, C16, ldc16, M, Nw, nT);
          break;
        case 2:
          gnn_wmma_gemm<2, 0><<<blocks, 256, 0, stream>>>(
              A1, lda1, W1, nullptr, 0, nullptr, biasPk, act, C, ldc, C16, ldc16, M, Nw, nT);
          break;
        case 4:
          gnn_wmma_gemm<4, 0><<<blocks, 256, 0, stream>>>(
              A1, lda1, W1, nullptr, 0, nullptr, biasPk, act, C, ldc, C16, ldc16, M, Nw, nT);
          break;
        default:
          gnn_wmma_gemm<6, 0><<<blocks, 256, 0, stream>>>(
              A1, lda1, W1, nullptr, 0, nullptr, biasPk, act, C, ldc, C16, ldc16, M, Nw, nT);
          break;
      }
    } else {
      gnn_wmma_gemm<6, 6><<<blocks, 256, 0, stream>>>(
          A1, lda1, W1, A2, lda2, W2, biasPk, act, C, ldc, C16, ldc16, M, Nw, nT);
    }
  };
  const _Float16* NOA = nullptr;

  // input conversion (zero-padded to K=32)
  {
    const int tot = Nn * 32;
    gnn_cvt_in<<<(tot + 255) / 256, 256, 0, stream>>>(xc, xc16, Nn, 32, 32);
    gnn_cvt_in<<<(tot + 255) / 256, 256, 0, stream>>>(xp, xp16, Nn, 16, 32);
    gnn_cvt_in<<<(tot + 255) / 256, 256, 0, stream>>>(xs, xs16, Nn, 24, 32);
  }

  // encoders -> hA (N x 192 concat)
  gemm(xc16, 32, wEC1, ktEC1, NOA, 0, nullptr, 0, bEC1, 1, nullptr, 0, t16, 128, Nn, 128);
  gemm(t16, 128, wEC2, ktEC2, NOA, 0, nullptr, 0, bEC2, 0, nullptr, 0, hA + 0,   H, Nn, 64);
  gemm(xp16, 32, wEP1, ktEP1, NOA, 0, nullptr, 0, bEP1, 1, nullptr, 0, t16, 128, Nn, 128);
  gemm(t16, 128, wEP2, ktEP2, NOA, 0, nullptr, 0, bEP2, 0, nullptr, 0, hA + 64,  H, Nn, 64);
  gemm(xs16, 32, wES1, ktES1, NOA, 0, nullptr, 0, bES1, 1, nullptr, 0, t16, 128, Nn, 128);
  gemm(t16, 128, wES2, ktES2, NOA, 0, nullptr, 0, bES2, 0, nullptr, 0, hA + 128, H, Nn, 64);

  // pre projection: hB = gelu(hA @ Wpre + b)
  gemm(hA, H, wPre, ktPre, NOA, 0, nullptr, 0, bPre, 1, nullptr, 0, hB, H, Nn, H);

  // SAGE conv layers
  _Float16* cur = hB;
  _Float16* nxt = hA;
  _Float16* wls[2] = {wWl0, wWl1};
  _Float16* wrs[2] = {wWr0, wWr1};
  int ktl[2] = {ktWl0, ktWl1}, ktr[2] = {ktWr0, ktWr1};
  float* bls[2] = {bBl0, bBl1};
  for (int l = 0; l < 2; ++l) {
    hipMemsetAsync(agg, 0, (size_t)Nn * H * 4, stream);
    hipMemsetAsync(deg, 0, (size_t)Nn * 4, stream);
    gnn_degree<<<(E + 255) / 256, 256, 0, stream>>>(dstI, deg, E);
    {
      const int tot = E * (H / 32);
      gnn_scatter<<<(tot + 255) / 256, 256, 0, stream>>>(cur, srcI, dstI, agg, E, H);
    }
    {
      const int tot = Nn * H;
      gnn_mean<<<(tot + 255) / 256, 256, 0, stream>>>(agg, deg, mean16, Nn, H);
    }
    // nxt = gelu(mean@Wl + bl + cur@Wr)
    gemm(mean16, H, wls[l], ktl[l], cur, H, wrs[l], ktr[l],
         bls[l], 1, nullptr, 0, nxt, H, Nn, H);
    _Float16* tmp = cur; cur = nxt; nxt = tmp;
  }

  // post: z (f32 -> d_out) + z16 (f16, reuse mean16)
  float* zout = (float*)d_out;
  _Float16* z16 = mean16;
  gemm(cur, H, wPost, ktPost, NOA, 0, nullptr, 0, bPost, 0, zout, H, z16, H, Nn, H);

  // decoders
  float* oC = zout + (size_t)Nn * H;
  float* oP = oC + (size_t)Nn * 32;
  float* oS = oP + (size_t)Nn * 16;
  gemm(z16 + 0, H, wDC1, ktDC1, NOA, 0, nullptr, 0, bDC1, 1, nullptr, 0, t16, 128, Nn, 128);
  gemm(t16, 128, wDC2, ktDC2, NOA, 0, nullptr, 0, bDC2, 0, oC, 32, nullptr, 0, Nn, 32);
  gemm(z16 + 64, H, wDP1, ktDP1, NOA, 0, nullptr, 0, bDP1, 1, nullptr, 0, t16, 128, Nn, 128);
  gemm(t16, 128, wDP2, ktDP2, NOA, 0, nullptr, 0, bDP2, 0, oP, 16, nullptr, 0, Nn, 16);
  gemm(z16 + 128, H, wDS1, ktDS1, NOA, 0, nullptr, 0, bDS1, 1, nullptr, 0, t16, 128, Nn, 128);
  gemm(t16, 128, wDS2, ktDS2, NOA, 0, nullptr, 0, bDS2, 0, oS, 24, nullptr, 0, Nn, 24);
}
